// NNConvHierarchicalModel_41291815584469
// MI455X (gfx1250) — matrix operand order repaired and checked by reading, not compiled, not run on hardware
//
#include <hip/hip_runtime.h>

typedef __attribute__((ext_vector_type(16))) _Float16 v16h;
typedef __attribute__((ext_vector_type(8)))  _Float16 v8h;
typedef __attribute__((ext_vector_type(8)))  float    v8f;

// ---------------------------------------------------------------------------
// W2 [32 x 1024] f32 -> f16 B-fragment layout for the outer-product GEMM:
//   msg = P @ W2r,  P[e, m*32+i] = h[e,m]*x[e,i],  W2r[m*32+i, o] = w2[m, i*32+o]
// K-chunk m, 16-col block nb (o<16 / o>=16).  B tile (32x16) layout:
//   lane n (0..15):  column n,  halves j hold K=i=j
//   lane n (16..31): column n-16, halves j hold K=i=16+j
// Stored chunk-major: out[(2m+nb)*512 + lane*16 + j].
// ---------------------------------------------------------------------------
__global__ void swizzle_w2_kernel(const float* __restrict__ w2,
                                  _Float16* __restrict__ out) {
  int tid = blockIdx.x * blockDim.x + threadIdx.x;
  if (tid >= 64 * 32 * 16) return;
  int j   = tid & 15;
  int l   = (tid >> 4) & 31;
  int blk = tid >> 9;
  int m   = blk >> 1;
  int nb  = blk & 1;
  int o   = nb * 16 + (l & 15);
  int i   = ((l >= 16) ? 16 : 0) + j;
  out[tid] = (_Float16)w2[m * 1024 + i * 32 + o];
}

// b2 [1024] viewed as B2[i][o]; two 16-col blocks, same B layout.
__global__ void swizzle_b2_kernel(const float* __restrict__ b2,
                                  _Float16* __restrict__ out) {
  int tid = blockIdx.x * blockDim.x + threadIdx.x;
  if (tid >= 2 * 32 * 16) return;
  int j  = tid & 15;
  int l  = (tid >> 4) & 31;
  int nb = tid >> 9;
  int o  = nb * 16 + (l & 15);
  int i  = ((l >= 16) ? 16 : 0) + j;
  out[tid] = (_Float16)b2[i * 32 + o];
}

static __device__ inline v16h splat_mul(v16h a, _Float16 s) {
  v16h r;
  #pragma unroll
  for (int i = 0; i < 16; ++i) r[i] = a[i] * s;
  return r;
}

static __device__ inline v8f wmma_f16(v16h a, v16h b, v8f c) {
  return __builtin_amdgcn_wmma_f32_16x16x32_f16(false, a, false, b,
                                                (short)0, c, false, false);
}

// ---------------------------------------------------------------------------
// Fused NNConv message kernel. One wave32 per 16-edge tile, 8 waves/block,
// grid-stride over tiles.  Per tile:
//   h   = relu(ef @ W1 + b1)             (VALU f32, 16 outputs/lane -> LDS)
//   acc = x_tile @ B2                    (2 WMMAs, bias term)
//   for m in 0..31 (unroll 4):
//        A_m = h[:,m] * x_frag           (ds_load_u16 bcast + v_pk_mul_f16)
//        acc = A_m @ W2r_m + acc         (2 WMMAs, D->C chain, no hazard NOPs)
//   atomicAdd msg rows into sum[dst], count into cnt[dst]
// W2/b2 staged once per block into dynamic LDS; unroll capped at 4 so the
// B-tile LDS loads stay inside the loop (full unroll made LICM hoist 2KB/lane
// of loop-invariant fragments -> scratch spills in round 2).
// ---------------------------------------------------------------------------
__global__ __launch_bounds__(256)
void edge_conv_msg_kernel(const float* __restrict__ x,
                          const int* __restrict__ src,
                          const int* __restrict__ dst,
                          const float* __restrict__ ef, int FE,
                          const float* __restrict__ w1,
                          const float* __restrict__ b1,
                          const _Float16* __restrict__ w2h,  // [64][32][16]
                          const _Float16* __restrict__ b2h,  // [2][32][16]
                          float* __restrict__ sum,
                          float* __restrict__ cnt,
                          int ntiles)
{
  extern __shared__ __align__(1024) char smem[];
  _Float16* w2s = (_Float16*)smem;               // 32768 halves = 64 KB
  _Float16* b2s = (_Float16*)(smem + 65536);     // 1024 halves  =  2 KB
  float*    w1s = (float*)(smem + 67584);        // 256 f32
  float*    b1s = (float*)(smem + 68608);        // 32 f32
  _Float16* hsm = (_Float16*)(smem + 68736);     // 8 waves x 16x32 halves = 8 KB

  int t = threadIdx.x;
  {
    const uint4* gw = (const uint4*)w2h;         // 4096 x 16B
    uint4* lw = (uint4*)w2s;
    for (int i = t; i < 4096; i += 256) lw[i] = gw[i];
    const uint4* gb = (const uint4*)b2h;         // 128 x 16B
    uint4* lb = (uint4*)b2s;
    if (t < 128) lb[t] = gb[t];
    if (t < FE * 32) w1s[t] = w1[t];
    if (t < 32)      b1s[t] = b1[t];
  }
  __syncthreads();

  int wid  = t >> 5;
  int l    = t & 31;
  int r    = l & 15;     // edge row within tile (M)
  int half = l >> 4;     // A-fragment K-split group / h output half
  int col  = l & 15;     // C/D column this lane holds
  _Float16* hrow = hsm + wid * 512 + r * 32;     // h row for this lane's edge

  for (int tile = blockIdx.x * 8 + wid; tile < ntiles; tile += gridDim.x * 8) {
    int e0 = tile * 16;
    int e  = e0 + r;

    // x A-fragment straight from global: lane needs K = {h16..h16+15}
    union UA { v16h v; v8h h[2]; } Ax;
    {
      int s = src[e];
      const float* xp = x + (size_t)s * 32;
      #pragma unroll
      for (int j = 0; j < 8; ++j) {
        Ax.h[0][j] = (_Float16)xp[half * 16 + j];
        Ax.h[1][j] = (_Float16)xp[half * 16 + 8 + j];
      }
    }

    // h = relu(ef @ W1 + b1): this lane computes outputs [half*16, half*16+16) -> LDS
    {
      float efr[8];
      const float* ep = ef + (size_t)e * FE;
      for (int k = 0; k < FE; ++k) efr[k] = ep[k];
      v16h hv;
      #pragma unroll
      for (int o2 = 0; o2 < 16; ++o2) {
        int o = half * 16 + o2;
        float a = b1s[o];
        for (int k = 0; k < FE; ++k) a += efr[k] * w1s[k * 32 + o];
        hv[o2] = (_Float16)(a > 0.f ? a : 0.f);
      }
      *(v16h*)(hrow + half * 16) = hv;
    }
    // wave-local store -> cross-lane load ordering (same wave, DS is in-order)
    asm volatile("s_wait_dscnt 0" ::: "memory");

    // bias term: acc = x @ B2
    v8f zero;
    #pragma unroll
    for (int i = 0; i < 8; ++i) zero[i] = 0.f;
    v16h Bb0 = *(const v16h*)(b2s + l * 16);
    v16h Bb1 = *(const v16h*)(b2s + 512 + l * 16);
    v8f accA = wmma_f16(Ax.v, Bb0, zero);
    v8f accB = wmma_f16(Ax.v, Bb1, zero);

    // K=1024 outer-product GEMM, accumulating in the WMMA C operand
    #pragma unroll 4
    for (int m = 0; m < 32; ++m) {
      _Float16 hv = hrow[m];                                     // 2B broadcast
      v16h B0 = *(const v16h*)(w2s + (2 * m) * 512 + l * 16);
      v16h B1 = *(const v16h*)(w2s + (2 * m + 1) * 512 + l * 16);
      v16h Am = splat_mul(Ax.v, hv);
      accA = wmma_f16(Am, B0, accA);
      accB = wmma_f16(Am, B1, accB);
    }

    // C layout: lane<16 -> col, rows M=reg; lane>=16 -> col, rows M=reg+8
    #pragma unroll
    for (int reg = 0; reg < 8; ++reg) {
      int erow = reg + (half ? 8 : 0);
      int d = dst[e0 + erow];
      atomicAdd(&sum[(size_t)d * 32 + col],      accA[reg]);
      atomicAdd(&sum[(size_t)d * 32 + 16 + col], accB[reg]);
    }
    if (l < 16) atomicAdd(&cnt[dst[e0 + l]], 1.0f);
  }
}

// ---------------------------------------------------------------------------
// out = relu(sum/max(cnt,1) + x @ RW + RB)   (one thread per row; f32)
// ---------------------------------------------------------------------------
__global__ void finalize_kernel(const float* __restrict__ sum,
                                const float* __restrict__ cnt,
                                const float* __restrict__ xin,
                                const float* __restrict__ rw,
                                const float* __restrict__ rb,
                                float* __restrict__ out, int n)
{
  int v = blockIdx.x * blockDim.x + threadIdx.x;
  if (v >= n) return;
  float xr[32];
  #pragma unroll
  for (int i = 0; i < 32; ++i) xr[i] = xin[(size_t)v * 32 + i];
  float c = cnt[v]; c = c > 1.f ? c : 1.f;
  float inv = 1.f / c;
  for (int o = 0; o < 32; ++o) {
    float a = sum[(size_t)v * 32 + o] * inv + rb[o];
    #pragma unroll
    for (int i = 0; i < 32; ++i) a += xr[i] * rw[i * 32 + o];
    out[(size_t)v * 32 + o] = a > 0.f ? a : 0.f;
  }
}

// out = x @ W + b    (32x32, one thread per row; f32)
__global__ void linear32_kernel(const float* __restrict__ xin,
                                const float* __restrict__ w,
                                const float* __restrict__ b,
                                float* __restrict__ out, int n)
{
  int v = blockIdx.x * blockDim.x + threadIdx.x;
  if (v >= n) return;
  float xr[32];
  #pragma unroll
  for (int i = 0; i < 32; ++i) xr[i] = xin[(size_t)v * 32 + i];
  for (int o = 0; o < 32; ++o) {
    float a = b[o];
    #pragma unroll
    for (int i = 0; i < 32; ++i) a += xr[i] * w[i * 32 + o];
    out[(size_t)v * 32 + o] = a;
  }
}

// scatter-add rows of m (indexed by srcIds) into sum[dstIds], count once per a
__global__ void scatter_kernel(const float* __restrict__ m,
                               const int* __restrict__ srcIds,
                               const int* __restrict__ dstIds,
                               float* __restrict__ sum,
                               float* __restrict__ cnt, int nA)
{
  int idx = blockIdx.x * blockDim.x + threadIdx.x;
  if (idx >= nA * 32) return;
  int a = idx >> 5, o = idx & 31;
  int si = srcIds[a], di = dstIds[a];
  atomicAdd(&sum[(size_t)di * 32 + o], m[(size_t)si * 32 + o]);
  if (o == 0) atomicAdd(&cnt[di], 1.0f);
}

// out = base + sum/max(cnt,1)
__global__ void add_mean_kernel(const float* __restrict__ base,
                                const float* __restrict__ sum,
                                const float* __restrict__ cnt,
                                float* __restrict__ out, int n)
{
  int idx = blockIdx.x * blockDim.x + threadIdx.x;
  if (idx >= n * 32) return;
  int v = idx >> 5;
  float c = cnt[v]; c = c > 1.f ? c : 1.f;
  out[idx] = base[idx] + sum[idx] / c;
}

// ---------------------------------------------------------------------------
extern "C" void kernel_launch(void* const* d_in, const int* in_sizes, int n_in,
                              void* d_out, int out_size, void* d_ws, size_t ws_size,
                              hipStream_t stream)
{
  (void)n_in; (void)out_size; (void)ws_size;
  const float* node_features   = (const float*)d_in[0];
  const int*   edge_index      = (const int*)d_in[1];
  const float* edge_features   = (const float*)d_in[2];
  const float* clique_features = (const float*)d_in[3];
  const int*   n2c_index       = (const int*)d_in[4];
  const int*   cedge_index     = (const int*)d_in[5];
  const float* cedge_features  = (const float*)d_in[6];
  const float* nn1_w  = (const float*)d_in[7];
  const float* nn1_b  = (const float*)d_in[8];
  const float* nn2_w  = (const float*)d_in[9];
  const float* nn2_b  = (const float*)d_in[10];
  const float* root_w = (const float*)d_in[11];
  const float* root_b = (const float*)d_in[12];
  const float* n2c_w  = (const float*)d_in[13];
  const float* n2c_b  = (const float*)d_in[14];
  const float* cnn1_w = (const float*)d_in[15];
  const float* cnn1_b = (const float*)d_in[16];
  const float* cnn2_w = (const float*)d_in[17];
  const float* cnn2_b = (const float*)d_in[18];
  const float* croot_w= (const float*)d_in[19];
  const float* croot_b= (const float*)d_in[20];
  const float* c2n_w  = (const float*)d_in[21];
  const float* c2n_b  = (const float*)d_in[22];

  const int N   = in_sizes[0] / 32;
  const int E   = in_sizes[1] / 2;
  const int FE  = in_sizes[2] / E;
  const int NC  = in_sizes[3] / 32;
  const int A   = in_sizes[4] / 2;
  const int EC  = in_sizes[5] / 2;
  const int FCE = in_sizes[6] / EC;
  const int L   = in_sizes[8] / 32;

  // workspace partition (256B-aligned bumps)
  char* p = (char*)d_ws;
  auto alloc = [&](size_t bytes) -> char* {
    char* r = p; p += (bytes + 255) & ~(size_t)255; return r;
  };
  float* xA    = (float*)alloc((size_t)N  * 32 * 4);
  float* xB    = (float*)alloc((size_t)N  * 32 * 4);
  float* cA    = (float*)alloc((size_t)NC * 32 * 4);
  float* cB    = (float*)alloc((size_t)NC * 32 * 4);
  float* nsum  = (float*)alloc((size_t)N  * 32 * 4);
  float* ncnt  = (float*)alloc((size_t)N  * 4);
  float* csum  = (float*)alloc((size_t)NC * 32 * 4);
  float* ccnt  = (float*)alloc((size_t)NC * 4);
  float* mbuf  = (float*)alloc((size_t)N  * 32 * 4);
  float* mcbuf = (float*)alloc((size_t)NC * 32 * 4);
  _Float16* w2h  = (_Float16*)alloc(64 * 32 * 16 * 2);
  _Float16* b2h  = (_Float16*)alloc(2 * 32 * 16 * 2);
  _Float16* cw2h = (_Float16*)alloc(64 * 32 * 16 * 2);
  _Float16* cb2h = (_Float16*)alloc(2 * 32 * 16 * 2);

  hipMemcpyAsync(xA, node_features,   (size_t)N  * 32 * 4, hipMemcpyDeviceToDevice, stream);
  hipMemcpyAsync(cA, clique_features, (size_t)NC * 32 * 4, hipMemcpyDeviceToDevice, stream);

  const int ntilesE  = E  / 16;   // E, EC are multiples of 16 in this problem
  const int ntilesEC = EC / 16;
  const size_t smem_bytes = 76928;          // 64K w2 + 2K b2 + w1 + b1 + 8K h
  int blocksE  = (ntilesE  + 7) / 8; if (blocksE  > 304) blocksE  = 304;
  int blocksEC = (ntilesEC + 7) / 8; if (blocksEC > 304) blocksEC = 304;

  for (int l = 0; l < L; ++l) {
    // ---- node-level NNConv ------------------------------------------------
    swizzle_w2_kernel<<<(32768 + 255) / 256, 256, 0, stream>>>(nn2_w + (size_t)l * 32 * 1024, w2h);
    swizzle_b2_kernel<<<(1024 + 255) / 256, 256, 0, stream>>>(nn2_b + (size_t)l * 1024, b2h);
    hipMemsetAsync(nsum, 0, (size_t)N * 32 * 4, stream);
    hipMemsetAsync(ncnt, 0, (size_t)N * 4, stream);
    edge_conv_msg_kernel<<<blocksE, 256, smem_bytes, stream>>>(
        xA, edge_index, edge_index + E, edge_features, FE,
        nn1_w + (size_t)l * FE * 32, nn1_b + (size_t)l * 32,
        w2h, b2h, nsum, ncnt, ntilesE);
    finalize_kernel<<<(N + 255) / 256, 256, 0, stream>>>(
        nsum, ncnt, xA, root_w + (size_t)l * 1024, root_b + (size_t)l * 32, xB, N);

    // ---- Node2Clique ------------------------------------------------------
    linear32_kernel<<<(N + 255) / 256, 256, 0, stream>>>(
        xB, n2c_w + (size_t)l * 1024, n2c_b + (size_t)l * 32, mbuf, N);
    hipMemsetAsync(csum, 0, (size_t)NC * 32 * 4, stream);
    hipMemsetAsync(ccnt, 0, (size_t)NC * 4, stream);
    scatter_kernel<<<((size_t)A * 32 + 255) / 256, 256, 0, stream>>>(
        mbuf, n2c_index, n2c_index + A, csum, ccnt, A);
    add_mean_kernel<<<((size_t)NC * 32 + 255) / 256, 256, 0, stream>>>(
        cA, csum, ccnt, cB, NC);

    // ---- clique-level NNConv ---------------------------------------------
    swizzle_w2_kernel<<<(32768 + 255) / 256, 256, 0, stream>>>(cnn2_w + (size_t)l * 32 * 1024, cw2h);
    swizzle_b2_kernel<<<(1024 + 255) / 256, 256, 0, stream>>>(cnn2_b + (size_t)l * 1024, cb2h);
    hipMemsetAsync(csum, 0, (size_t)NC * 32 * 4, stream);
    hipMemsetAsync(ccnt, 0, (size_t)NC * 4, stream);
    edge_conv_msg_kernel<<<blocksEC, 256, smem_bytes, stream>>>(
        cB, cedge_index, cedge_index + EC, cedge_features, FCE,
        cnn1_w + (size_t)l * FCE * 32, cnn1_b + (size_t)l * 32,
        cw2h, cb2h, csum, ccnt, ntilesEC);
    finalize_kernel<<<(NC + 255) / 256, 256, 0, stream>>>(
        csum, ccnt, cB, croot_w + (size_t)l * 1024, croot_b + (size_t)l * 32, cA, NC);

    // ---- Clique2Node ------------------------------------------------------
    linear32_kernel<<<(NC + 255) / 256, 256, 0, stream>>>(
        cA, c2n_w + (size_t)l * 1024, c2n_b + (size_t)l * 32, mcbuf, NC);
    hipMemsetAsync(nsum, 0, (size_t)N * 32 * 4, stream);
    hipMemsetAsync(ncnt, 0, (size_t)N * 4, stream);
    scatter_kernel<<<((size_t)A * 32 + 255) / 256, 256, 0, stream>>>(
        mcbuf, n2c_index + A, n2c_index, nsum, ncnt, A);
    add_mean_kernel<<<((size_t)N * 32 + 255) / 256, 256, 0, stream>>>(
        xB, nsum, ncnt, xA, N);
  }

  float* out = (float*)d_out;
  hipMemcpyAsync(out,                  xA, (size_t)N  * 32 * 4, hipMemcpyDeviceToDevice, stream);
  hipMemcpyAsync(out + (size_t)N * 32, cA, (size_t)NC * 32 * 4, hipMemcpyDeviceToDevice, stream);
}
